// MoELayer_63556926046582
// MI455X (gfx1250) — matrix operand order repaired and checked by reading, not compiled, not run on hardware
//
#include <hip/hip_runtime.h>
#include <hip/hip_bf16.h>

typedef __attribute__((ext_vector_type(16))) __bf16 bf16x16;
typedef __attribute__((ext_vector_type(8)))  float  f32x8;
typedef int v4i32 __attribute__((vector_size(16)));

#define DIM   1024
#define HEADS 16
#define HEADD 64
#define SEQ   1024
#define SEQ2  2048
#define NEXP  16
#define EXPD  512
#define DSH   2048
#define TOPK  4

#if defined(__gfx1250__) && __has_builtin(__builtin_amdgcn_global_load_async_to_lds_b128) && __has_builtin(__builtin_amdgcn_s_wait_asynccnt)
#define ASYNC_COPY 1
#else
#define ASYNC_COPY 0
#endif

// ---------------------------------------------------------------------------
// WMMA helpers (CDNA5 16x16x32 bf16, fp32 accumulate), wave32.
// LDS operand tiles are stored in *fragment order* so each lane's 16 bf16 are
// contiguous + 16B aligned: fragment load = 2x ds_load_b128.
//   A layout per 32-K slab: row m, K permuted by
//     a_perm(k) = (k&7) | ((k&8)<<1) | ((k&16)>>1)           (ISA 7.12.2)
//   B layout: [n][k], k contiguous (lane<16 -> K 0-7,16-23, lane>=16 -> K 8-15,24-31)
// ---------------------------------------------------------------------------
struct u4x2 { uint4 a, b; };
struct bfpair { __bf16 lo, hi; };

__device__ __forceinline__ f32x8 wmma_bf16(bf16x16 a, bf16x16 b, f32x8 c) {
  return __builtin_amdgcn_wmma_f32_16x16x32_bf16(false, a, false, b, (short)0, c,
                                                 false, false);
}

__device__ __forceinline__ bf16x16 frag_from(const uint4* p) {
  u4x2 u;
  u.a = p[0];
  u.b = p[1];
  return __builtin_bit_cast(bf16x16, u);
}

// rs = row stride in uint4 units (bf16 stride / 8)
__device__ __forceinline__ bf16x16 load_a_frag(const uint4* base, int m0, int rs) {
  int lane = threadIdx.x & 31;
  return frag_from(base + (size_t)(m0 + (lane & 15)) * rs + (lane >> 4) * 2);
}
__device__ __forceinline__ bf16x16 load_b_frag(const uint4* base, int n0, int rs) {
  int lane = threadIdx.x & 31;
  const uint4* p = base + (size_t)(n0 + (lane & 15)) * rs + (lane >> 4);
  u4x2 u;
  u.a = p[0];
  u.b = p[2];
  return __builtin_bit_cast(bf16x16, u);
}

__device__ __forceinline__ int a_perm(int k) {
  return (k & 7) | ((k & 8) << 1) | ((k & 16) >> 1);
}
__device__ __forceinline__ int perm8(int kb) {  // a_perm for 8-aligned block start
  return ((kb & 8) << 1) | ((kb & 16) >> 1);
}

__device__ __forceinline__ unsigned pack_bf2(float a, float b) {
  bfpair p{(__bf16)a, (__bf16)b};  // lowers to v_cvt_pk_bf16_f32
  return __builtin_bit_cast(unsigned, p);
}

// async global->LDS 16B copy (ASYNCcnt path); fallback: LDS store via VGPRs
__device__ __forceinline__ void copy16(const void* g, void* l) {
#if ASYNC_COPY
  __builtin_amdgcn_global_load_async_to_lds_b128(
      (__attribute__((address_space(1))) v4i32*)(v4i32*)(void*)g,
      (__attribute__((address_space(3))) v4i32*)(v4i32*)l, 0, 0);
#else
  *(uint4*)l = *(const uint4*)g;
#endif
}
__device__ __forceinline__ void copy16_join() {
#if ASYNC_COPY
  __builtin_amdgcn_s_wait_asynccnt(0);
#endif
}

__device__ __forceinline__ float sigmoidf_fast(float x) {
  return 1.0f / (1.0f + __expf(-x));
}
__device__ __forceinline__ float siluf(float x) { return x * sigmoidf_fast(x); }

// ---------------------------------------------------------------------------
// K0: RMSNorm -> bf16
// ---------------------------------------------------------------------------
__global__ __launch_bounds__(256) void k_rmsnorm_bf16(const float* __restrict__ X,
                                                      const float* __restrict__ w,
                                                      __bf16* __restrict__ out) {
  int row = blockIdx.x;
  const float* x = X + (size_t)row * DIM;
  __shared__ float red[8];
  float ss = 0.f;
  for (int d = threadIdx.x; d < DIM; d += 256) {
    float v = x[d];
    ss += v * v;
  }
#pragma unroll
  for (int off = 16; off; off >>= 1) ss += __shfl_down(ss, off, 32);
  if ((threadIdx.x & 31) == 0) red[threadIdx.x >> 5] = ss;
  __syncthreads();
  if (threadIdx.x == 0) {
    float t = 0.f;
#pragma unroll
    for (int i = 0; i < 8; ++i) t += red[i];
    red[0] = rsqrtf(t / (float)DIM + 1e-5f);
  }
  __syncthreads();
  float r = red[0];
  for (int d = threadIdx.x; d < DIM; d += 256)
    out[(size_t)row * DIM + d] = (__bf16)(x[d] * r * w[d]);
}

// ---------------------------------------------------------------------------
// K1: generic GEMM  C[f32] = A[bf16, MxK] @ B[f32->bf16, KxN] (+ R)
// tile 128x64, K-step 32, 256 threads = 8 waves (4x2), each wave 32x32.
// B staging: thread converts a k-row pair -> packed bf16 dword -> ds_store_b32.
// ---------------------------------------------------------------------------
__global__ __launch_bounds__(256) void k_gemm_bf16(const __bf16* __restrict__ A,
                                                   int lda,
                                                   const float* __restrict__ B,
                                                   int ldb, float* __restrict__ C,
                                                   int ldc,
                                                   const float* __restrict__ R,
                                                   int Kdim) {
  __shared__ uint4 sA[128 * 5];  // 128 rows x 40 bf16, A fragment order
  __shared__ uint4 sB[64 * 5];   // 64 n x 40 bf16, [n][k] k-contiguous
  unsigned* sBu = (unsigned*)sB;
  int tid = threadIdx.x;
  int wave = tid >> 5, lane = tid & 31;
  int m_block = blockIdx.y * 128;
  int n_block = blockIdx.x * 64;
  int wm = (wave >> 1) * 32;
  int wn = (wave & 1) * 32;
  f32x8 acc[2][2] = {{}, {}};
  for (int k0 = 0; k0 < Kdim; k0 += 32) {
    {  // stage A 128x32: thread copies 16 contiguous bf16 (2 permuted 8-blocks)
      int r = tid >> 1, c0 = (tid & 1) * 16;
      const __bf16* src = A + (size_t)(m_block + r) * lda + k0 + c0;
      __bf16* dst = ((__bf16*)sA) + r * 40;
      uint4 v0 = *(const uint4*)src;
      uint4 v1 = *(const uint4*)(src + 8);
      *(uint4*)(dst + perm8(c0)) = v0;
      *(uint4*)(dst + perm8(c0 + 8)) = v1;
    }
    {  // stage B 32x64 fp32->bf16 transposed: k-row pair x 4 cols per thread
      int rp = tid >> 4;          // k-row pair 0..15 -> rows 2rp, 2rp+1
      int c0 = (tid & 15) * 4;    // 4 columns
      const float* s0 = B + (size_t)(k0 + 2 * rp) * ldb + n_block + c0;
      const float* s1 = s0 + ldb;
      float4 a0 = *(const float4*)s0;
      float4 a1 = *(const float4*)s1;
      sBu[(c0 + 0) * 20 + rp] = pack_bf2(a0.x, a1.x);
      sBu[(c0 + 1) * 20 + rp] = pack_bf2(a0.y, a1.y);
      sBu[(c0 + 2) * 20 + rp] = pack_bf2(a0.z, a1.z);
      sBu[(c0 + 3) * 20 + rp] = pack_bf2(a0.w, a1.w);
      if (k0 + 32 < Kdim)  // global_prefetch_b8 next B tile
        __builtin_prefetch(s0 + (size_t)32 * ldb, 0, 1);
    }
    __syncthreads();
    bf16x16 af0 = load_a_frag(sA, wm, 5);
    bf16x16 af1 = load_a_frag(sA, wm + 16, 5);
    bf16x16 bf0 = load_b_frag(sB, wn, 5);
    bf16x16 bf1 = load_b_frag(sB, wn + 16, 5);
    acc[0][0] = wmma_bf16(af0, bf0, acc[0][0]);
    acc[0][1] = wmma_bf16(af0, bf1, acc[0][1]);
    acc[1][0] = wmma_bf16(af1, bf0, acc[1][0]);
    acc[1][1] = wmma_bf16(af1, bf1, acc[1][1]);
    __syncthreads();
  }
  bool has_r = (R != nullptr);
#pragma unroll
  for (int mi = 0; mi < 2; ++mi)
#pragma unroll
    for (int ni = 0; ni < 2; ++ni) {
      int n = n_block + wn + ni * 16 + (lane & 15);
      int m = m_block + wm + mi * 16 + (lane >> 4) * 8;
#pragma unroll
      for (int i = 0; i < 8; ++i) {
        size_t idx = (size_t)(m + i) * ldc + n;
        float v = acc[mi][ni][i];
        if (has_r) v += R[idx];
        C[idx] = v;
      }
    }
}

// ---------------------------------------------------------------------------
// K2: RoPE + dual-interleave.  QKV f32 [2048][3072] ->
//   Q  bf16 [h][2048][64]  (rope)
//   K  bf16 [h][2048][64]  (rope)         -> natural layout == B-frag order
//   Vt bf16 [h][64][2048]  (transposed)   -> B-frag order for P@V
// interleaved index sq = 2*s + c with source row = c*1024 + s
// ---------------------------------------------------------------------------
__global__ __launch_bounds__(512) void k_rope(const float* __restrict__ QKV,
                                              __bf16* __restrict__ Q,
                                              __bf16* __restrict__ K,
                                              __bf16* __restrict__ Vt) {
  int row = blockIdx.x;  // c*1024 + s
  int c = row >> 10, s = row & 1023;
  int sq = 2 * s + c;
  int t = threadIdx.x;  // 16 heads * 32 pairs
  int h = t >> 5, j = t & 31;
  const float* base = QKV + (size_t)row * 3072;
  float inv = __powf(10000.0f, -(2.0f * (float)j) / 64.0f);
  float fr = (float)s * inv;
  float cs = __cosf(fr), sn = __sinf(fr);
  float q1 = base[h * 64 + j], q2 = base[h * 64 + j + 32];
  float k1 = base[1024 + h * 64 + j], k2 = base[1024 + h * 64 + j + 32];
  float v1 = base[2048 + h * 64 + j], v2 = base[2048 + h * 64 + j + 32];
  __bf16* qrow = Q + ((size_t)h * SEQ2 + sq) * 64;
  qrow[j] = (__bf16)(q1 * cs + q2 * sn);
  qrow[j + 32] = (__bf16)(-q1 * sn + q2 * cs);
  __bf16* krow = K + ((size_t)h * SEQ2 + sq) * 64;
  krow[j] = (__bf16)(k1 * cs + k2 * sn);
  krow[j + 32] = (__bf16)(-k1 * sn + k2 * cs);
  __bf16* vb = Vt + (size_t)h * 64 * SEQ2;
  vb[(size_t)j * SEQ2 + sq] = (__bf16)v1;
  vb[(size_t)(j + 32) * SEQ2 + sq] = (__bf16)v2;
}

// ---------------------------------------------------------------------------
// K3: causal flash attention, one (head, 128-query tile) per block.
// 8 waves; wave w owns query rows [w*16, w*16+16). Online softmax via LDS.
// K/V tiles staged with async global->LDS copies (ASYNCcnt).
// ---------------------------------------------------------------------------
__global__ __launch_bounds__(256) void k_attn(const __bf16* __restrict__ Q,
                                              const __bf16* __restrict__ K,
                                              const __bf16* __restrict__ Vt,
                                              __bf16* __restrict__ O) {
  int h = blockIdx.x, qt = blockIdx.y;
  int q0 = qt * 128;
  int tid = threadIdx.x, wave = tid >> 5, lane = tid & 31;
  __shared__ uint4 sQ[128 * 9];  // 128 rows x (2 slabs x 32 + 8 pad), A frag order
  __shared__ uint4 sK[32 * 9];   // 32 keys x (64 d + 8 pad), natural = B order
  __shared__ uint4 sV[64 * 5];   // 64 d x (32 keys + 8 pad), Vt tile = B order
  __shared__ float sS[8][16][32];
  __shared__ uint4 sP[8 * 16 * 5];  // per wave: 16 rows x 40, A frag order
  __shared__ float sAlpha[8][16];
  __shared__ float sM[8][16], sL[8][16];
  __bf16* sQb = (__bf16*)sQ;
  __bf16* sKb = (__bf16*)sK;
  __bf16* sVb = (__bf16*)sV;
  __bf16* sPb = (__bf16*)sP;
  {  // stage Q tile 128x64 (2 K-slabs, permuted blocks)
    int r = tid >> 1, c0 = (tid & 1) * 32;
    const __bf16* src = Q + ((size_t)h * SEQ2 + q0 + r) * 64 + c0;
    __bf16* dst = sQb + r * 72 + c0;  // c0 = slab*32
    uint4 v0 = *(const uint4*)src;
    uint4 v1 = *(const uint4*)(src + 8);
    uint4 v2 = *(const uint4*)(src + 16);
    uint4 v3 = *(const uint4*)(src + 24);
    *(uint4*)(dst + perm8(0)) = v0;
    *(uint4*)(dst + perm8(8)) = v1;
    *(uint4*)(dst + perm8(16)) = v2;
    *(uint4*)(dst + perm8(24)) = v3;
  }
  if (lane < 16) {
    sM[wave][lane] = -1e30f;
    sL[wave][lane] = 0.f;
  }
  __syncthreads();
  bf16x16 qf0 = load_a_frag(sQ, wave * 16, 9);      // d slab 0
  bf16x16 qf1 = load_a_frag(sQ + 4, wave * 16, 9);  // d slab 1
  f32x8 of[4] = {};
  int kend = q0 + 128;  // causal: keys <= max query row in tile
  for (int kt0 = 0; kt0 < kend; kt0 += 32) {
    {  // stage K tile [key=32][d=64] : one async b128 per thread
      int r = tid >> 3, c0 = (tid & 7) * 8;
      copy16(K + ((size_t)h * SEQ2 + kt0 + r) * 64 + c0, sKb + r * 72 + c0);
    }
    {  // stage V tile [d=64][key=32] : one async b128 per thread
      int d = tid >> 2, c0 = (tid & 3) * 8;
      copy16(Vt + ((size_t)h * 64 + d) * SEQ2 + kt0 + c0, sVb + d * 40 + c0);
    }
    copy16_join();
    __syncthreads();
    // S[16x32] = Q[16x64] @ K-tile^T, two K-dim slabs of 32
    f32x8 sf0 = {}, sf1 = {};
    sf0 = wmma_bf16(qf0, load_b_frag(sK, 0, 9), sf0);
    sf1 = wmma_bf16(qf0, load_b_frag(sK, 16, 9), sf1);
    sf0 = wmma_bf16(qf1, load_b_frag(sK + 4, 0, 9), sf0);
    sf1 = wmma_bf16(qf1, load_b_frag(sK + 4, 16, 9), sf1);
    {  // spill S (pre-scaled) to LDS
      int n = lane & 15, mr = (lane >> 4) * 8;
#pragma unroll
      for (int i = 0; i < 8; ++i) {
        sS[wave][mr + i][n] = sf0[i] * 0.125f;
        sS[wave][mr + i][n + 16] = sf1[i] * 0.125f;
      }
    }
    __syncthreads();
    if (lane < 16) {  // online softmax, one row per lane; P written A-frag order
      int r = lane;
      int qrow = q0 + wave * 16 + r;
      float mo = sM[wave][r], lo = sL[wave][r];
      float mx = mo;
      for (int j = 0; j < 32; ++j) {
        float s = (kt0 + j > qrow) ? -1e30f : sS[wave][r][j];
        mx = fmaxf(mx, s);
      }
      float sum = 0.f;
      __bf16* prow = sPb + (wave * 16 + r) * 40;
      for (int j = 0; j < 32; ++j) {
        float p = (kt0 + j > qrow) ? 0.f : __expf(sS[wave][r][j] - mx);
        prow[a_perm(j)] = (__bf16)p;
        sum += p;
      }
      float alpha = __expf(mo - mx);
      sAlpha[wave][r] = alpha;
      sM[wave][r] = mx;
      sL[wave][r] = lo * alpha + sum;
    }
    __syncthreads();
    {  // rescale running O
      int mr = (lane >> 4) * 8;
#pragma unroll
      for (int i = 0; i < 8; ++i) {
        float a = sAlpha[wave][mr + i];
        of[0][i] *= a; of[1][i] *= a; of[2][i] *= a; of[3][i] *= a;
      }
    }
    // O += P[16x32] @ V[32x64]
    bf16x16 pf = load_a_frag(sP + wave * 16 * 5, 0, 5);
#pragma unroll
    for (int ni = 0; ni < 4; ++ni)
      of[ni] = wmma_bf16(pf, load_b_frag(sV, ni * 16, 5), of[ni]);
    __syncthreads();
  }
  {  // normalize + de-interleave store: sq -> row (sq&1)*1024 + (sq>>1)
    int mr = (lane >> 4) * 8, n0 = lane & 15;
#pragma unroll
    for (int i = 0; i < 8; ++i) {
      int r = wave * 16 + mr + i;
      int sq = q0 + r;
      float inv_l = 1.0f / sL[wave][mr + i];
      int orow = (sq & 1) * SEQ + (sq >> 1);
      __bf16* dst = O + (size_t)orow * DIM + h * 64;
#pragma unroll
      for (int ni = 0; ni < 4; ++ni)
        dst[ni * 16 + n0] = (__bf16)(of[ni][i] * inv_l);
    }
  }
}

// ---------------------------------------------------------------------------
// K4: router. One block per token.
// ---------------------------------------------------------------------------
__global__ __launch_bounds__(256) void k_router(
    const __bf16* __restrict__ Xffn, const float* __restrict__ pkeys,
    const float* __restrict__ fkeys, const int* __restrict__ pidx,
    const int* __restrict__ fidx, const float* __restrict__ pval,
    const float* __restrict__ fval, const float* __restrict__ pbias,
    const float* __restrict__ fbias, float* __restrict__ sc) {
  int row = blockIdx.x;  // 0..2047
  int gr = row >> 10, tloc = row & 1023;
  const float* keys = gr ? fkeys : pkeys;
  const int* idx = (gr ? fidx : pidx) + tloc * TOPK;
  const float* val = (gr ? fval : pval) + tloc * TOPK;
  const float* bias = gr ? fbias : pbias;
  __shared__ float part[256];
  __shared__ float logits[NEXP];
  int t = threadIdx.x;
  int e = t & 15, slot = t >> 4;
  float s = 0.f;
  for (int d = slot; d < DIM; d += 16)
    s += (float)Xffn[(size_t)row * DIM + d] * keys[d * NEXP + e];
  part[t] = s;
  __syncthreads();
  if (t < NEXP) {
    float tot = 0.f;
#pragma unroll
    for (int i = 0; i < 16; ++i) tot += part[i * 16 + t];
    logits[t] = tot;
  }
  __syncthreads();
  if (t == 0) {
    float g[TOPK];
    float sum = 0.f;
#pragma unroll
    for (int k = 0; k < TOPK; ++k) {
      int e2 = idx[k];
      g[k] = sigmoidf_fast(val[k] + logits[e2] + bias[e2]);
      sum += g[k];
    }
#pragma unroll
    for (int k = 0; k < TOPK; ++k) sc[row * TOPK + k] = g[k] / sum;
  }
}

// ---------------------------------------------------------------------------
// K5: MoE grouped GEMM. Block = (64-token tile, expert, group).
// Dense over experts (reference semantics); per-token scale 0 if unselected.
// gate/up (K=1024) -> silu*u -> LDS H (A-frag order) -> down (K=512) ->
// scaled atomic accumulate into Y (holds residual + shared expert already).
// ---------------------------------------------------------------------------
__global__ __launch_bounds__(256) void k_moe(const __bf16* __restrict__ Xffn,
                                             const float* __restrict__ Wp,
                                             const float* __restrict__ Wf,
                                             const int* __restrict__ pidx,
                                             const int* __restrict__ fidx,
                                             const float* __restrict__ sc,
                                             float* __restrict__ Y) {
  int tile = blockIdx.x, e = blockIdx.y, gr = blockIdx.z;
  int row0 = gr * SEQ + tile * 64;
  const float* W = gr ? Wf : Wp;  // [3][16][1024][512]
  const int* idx = gr ? fidx : pidx;
  const float* Wg = W + ((size_t)0 * NEXP + e) * DIM * EXPD;
  const float* Wu = W + ((size_t)1 * NEXP + e) * DIM * EXPD;
  const float* Wd = W + ((size_t)2 * NEXP + e) * DIM * EXPD;
  __shared__ float sScale[64];
  __shared__ uint4 sX[64 * 5];   // A frag order, 64 rows x 40
  __shared__ uint4 sW0[64 * 5];  // B order [n][k]
  __shared__ uint4 sW1[64 * 5];
  __shared__ uint4 sH[64 * 65];  // 64 rows x (16 slabs x 32 + 8 pad), A frag order
  __bf16* sXb = (__bf16*)sX;
  unsigned* sW0u = (unsigned*)sW0;
  unsigned* sW1u = (unsigned*)sW1;
  __bf16* sW0b = (__bf16*)sW0;
  __bf16* sHb = (__bf16*)sH;
  int tid = threadIdx.x, wave = tid >> 5, lane = tid & 31;
  if (tid < 64) {
    int tloc = tile * 64 + tid;
    float scv = 0.f;
#pragma unroll
    for (int k = 0; k < TOPK; ++k)
      if (idx[tloc * TOPK + k] == e) scv = sc[(size_t)(gr * SEQ + tloc) * TOPK + k];
    sScale[tid] = scv;
  }
  __syncthreads();
  int wm = (wave >> 1) * 16;  // 4 waves over 64 rows
  int wn = (wave & 1) * 32;   // 2 waves over 64 cols
  // ---- phase 1: gate & up, f-tiles of 64 ----
  for (int ft = 0; ft < EXPD; ft += 64) {
    f32x8 cg[2] = {}, cu[2] = {};
    for (int k0 = 0; k0 < DIM; k0 += 32) {
      {  // stage X 64x32 (one permuted 8-block per thread)
        int r = tid >> 2, c0 = (tid & 3) * 8;
        const __bf16* src = Xffn + (size_t)(row0 + r) * DIM + k0 + c0;
        uint4 v = *(const uint4*)src;
        *(uint4*)(sXb + r * 40 + perm8(c0)) = v;
      }
      {  // stage Wg/Wu 32x64 fp32->bf16 transposed: k-row pair x 4 cols
        int rp = tid >> 4, c0 = (tid & 15) * 4;
        const float* g0 = Wg + (size_t)(k0 + 2 * rp) * EXPD + ft + c0;
        const float* u0 = Wu + (size_t)(k0 + 2 * rp) * EXPD + ft + c0;
        float4 ga = *(const float4*)g0;
        float4 gb = *(const float4*)(g0 + EXPD);
        float4 ua = *(const float4*)u0;
        float4 ub = *(const float4*)(u0 + EXPD);
        sW0u[(c0 + 0) * 20 + rp] = pack_bf2(ga.x, gb.x);
        sW0u[(c0 + 1) * 20 + rp] = pack_bf2(ga.y, gb.y);
        sW0u[(c0 + 2) * 20 + rp] = pack_bf2(ga.z, gb.z);
        sW0u[(c0 + 3) * 20 + rp] = pack_bf2(ga.w, gb.w);
        sW1u[(c0 + 0) * 20 + rp] = pack_bf2(ua.x, ub.x);
        sW1u[(c0 + 1) * 20 + rp] = pack_bf2(ua.y, ub.y);
        sW1u[(c0 + 2) * 20 + rp] = pack_bf2(ua.z, ub.z);
        sW1u[(c0 + 3) * 20 + rp] = pack_bf2(ua.w, ub.w);
        if (k0 + 32 < DIM) {
          __builtin_prefetch(g0 + (size_t)32 * EXPD, 0, 1);
          __builtin_prefetch(u0 + (size_t)32 * EXPD, 0, 1);
        }
      }
      __syncthreads();
      bf16x16 af = load_a_frag(sX, wm, 5);
#pragma unroll
      for (int ni = 0; ni < 2; ++ni) {
        cg[ni] = wmma_bf16(af, load_b_frag(sW0, wn + ni * 16, 5), cg[ni]);
        cu[ni] = wmma_bf16(af, load_b_frag(sW1, wn + ni * 16, 5), cu[ni]);
      }
      __syncthreads();
    }
    {  // h = silu(g)*u -> sH in A-frag order (slab = f>>5, perm within slab)
      int n0 = lane & 15, mr = wm + (lane >> 4) * 8;
#pragma unroll
      for (int ni = 0; ni < 2; ++ni) {
        int f = ft + wn + ni * 16 + n0;
        int fo = (f >> 5) * 32 + a_perm(f & 31);
#pragma unroll
        for (int i = 0; i < 8; ++i)
          sHb[(size_t)(mr + i) * 520 + fo] = (__bf16)(siluf(cg[ni][i]) * cu[ni][i]);
      }
    }
    __syncthreads();
  }
  // ---- phase 2: down proj out[64][1024] = H[64][512] @ Wd^T ----
  for (int dt = 0; dt < DIM; dt += 64) {
    f32x8 cd[2] = {};
    for (int k0 = 0; k0 < EXPD; k0 += 32) {
      {  // stage B[n=d][k=f] = Wd[d][f]: contiguous in f -> vector LDS store
        int n = tid >> 2, kk0 = (tid & 3) * 8;
        const float* src = Wd + (size_t)(dt + n) * EXPD + k0 + kk0;
        __bf16 tmp[8];
#pragma unroll
        for (int i = 0; i < 8; ++i) tmp[i] = (__bf16)src[i];
        *(uint4*)(sW0b + n * 40 + kk0) = *(uint4*)tmp;
      }
      __syncthreads();
      bf16x16 af = load_a_frag(sH + (k0 >> 5) * 4, wm, 65);
#pragma unroll
      for (int ni = 0; ni < 2; ++ni)
        cd[ni] = wmma_bf16(af, load_b_frag(sW0, wn + ni * 16, 5), cd[ni]);
      __syncthreads();
    }
    {  // scaled atomic accumulate
      int n0 = lane & 15, mr = wm + (lane >> 4) * 8;
#pragma unroll
      for (int ni = 0; ni < 2; ++ni)
#pragma unroll
        for (int i = 0; i < 8; ++i) {
          float v = cd[ni][i] * sScale[mr + i];
          float* dst = &Y[(size_t)(row0 + mr + i) * DIM + dt + wn + ni * 16 + n0];
          __hip_atomic_fetch_add(dst, v, __ATOMIC_RELAXED, __HIP_MEMORY_SCOPE_AGENT);
        }
    }
  }
}

// ---------------------------------------------------------------------------
// K6: shared-expert SwiGLU combine: H[t][f] = silu(UP[t][f]) * UP[t][f+2048]
// ---------------------------------------------------------------------------
__global__ __launch_bounds__(256) void k_silu_mul(const float* __restrict__ UP,
                                                  __bf16* __restrict__ H) {
  size_t i = (size_t)blockIdx.x * 256 + threadIdx.x;  // 2048*2048
  size_t row = i >> 11, col = i & 2047;
  float a = UP[row * 4096 + col];
  float b = UP[row * 4096 + 2048 + col];
  H[i] = (__bf16)(siluf(a) * b);
}

// ---------------------------------------------------------------------------
extern "C" void kernel_launch(void* const* d_in, const int* in_sizes, int n_in,
                              void* d_out, int out_size, void* d_ws,
                              size_t ws_size, hipStream_t stream) {
  (void)in_sizes; (void)n_in; (void)out_size; (void)ws_size;
  const float* x_input   = (const float*)d_in[0];
  const int*   p_indices = (const int*)d_in[1];
  const float* p_values  = (const float*)d_in[2];
  const int*   f_indices = (const int*)d_in[3];
  const float* f_values  = (const float*)d_in[4];
  const float* attn_nw   = (const float*)d_in[5];
  const float* ffn_nw    = (const float*)d_in[6];
  const float* W_attn    = (const float*)d_in[7];
  const float* W_attn_o  = (const float*)d_in[8];
  const float* ffn_up    = (const float*)d_in[9];
  const float* ffn_down  = (const float*)d_in[10];
  const float* p_exp     = (const float*)d_in[11];
  const float* f_exp     = (const float*)d_in[12];
  const float* p_keys    = (const float*)d_in[13];
  const float* f_keys    = (const float*)d_in[14];
  const float* p_bias    = (const float*)d_in[15];
  const float* f_bias    = (const float*)d_in[16];
  float* Y = (float*)d_out;

  char* ws = (char*)d_ws;
  size_t off = 0;
  auto alloc = [&](size_t bytes) {
    size_t r = off;
    off = (off + bytes + 255) & ~(size_t)255;
    return r;
  };
  __bf16* Xbf   = (__bf16*)(ws + alloc((size_t)SEQ2 * DIM * 2));
  float*  QKV   = (float*)(ws + alloc((size_t)SEQ2 * 3 * DIM * 4));
  __bf16* Qb    = (__bf16*)(ws + alloc((size_t)HEADS * SEQ2 * HEADD * 2));
  __bf16* Kb    = (__bf16*)(ws + alloc((size_t)HEADS * SEQ2 * HEADD * 2));
  __bf16* Vtb   = (__bf16*)(ws + alloc((size_t)HEADS * SEQ2 * HEADD * 2));
  __bf16* Ob    = (__bf16*)(ws + alloc((size_t)SEQ2 * DIM * 2));
  float*  XFI   = (float*)(ws + alloc((size_t)SEQ2 * DIM * 4));  // x_ffn_input
  __bf16* Xffnb = (__bf16*)(ws + alloc((size_t)SEQ2 * DIM * 2));
  float*  SC    = (float*)(ws + alloc((size_t)SEQ2 * TOPK * 4));
  float*  UP    = (float*)(ws + alloc((size_t)SEQ2 * 2 * DSH * 4));
  __bf16* Hb    = (__bf16*)(ws + alloc((size_t)SEQ2 * DSH * 2));

  // 1) x = rmsnorm(x_input) -> bf16
  k_rmsnorm_bf16<<<SEQ2, 256, 0, stream>>>(x_input, attn_nw, Xbf);
  // 2) QKV = x @ W_attn (f32 out for RoPE precision)
  k_gemm_bf16<<<dim3(3 * DIM / 64, SEQ2 / 128), 256, 0, stream>>>(
      Xbf, DIM, W_attn, 3 * DIM, QKV, 3 * DIM, nullptr, DIM);
  // 3) RoPE + interleave
  k_rope<<<SEQ2, 512, 0, stream>>>(QKV, Qb, Kb, Vtb);
  // 4) attention (flash, causal), output de-interleaved bf16
  k_attn<<<dim3(HEADS, SEQ2 / 128), 256, 0, stream>>>(Qb, Kb, Vtb, Ob);
  // 5) x_ffn_input = O @ W_attn_o + x_input
  k_gemm_bf16<<<dim3(DIM / 64, SEQ2 / 128), 256, 0, stream>>>(
      Ob, DIM, W_attn_o, DIM, XFI, DIM, x_input, DIM);
  // 6) x_ffn = rmsnorm(x_ffn_input) -> bf16
  k_rmsnorm_bf16<<<SEQ2, 256, 0, stream>>>(XFI, ffn_nw, Xffnb);
  // 7) router scores
  k_router<<<SEQ2, 256, 0, stream>>>(Xffnb, p_keys, f_keys, p_indices, f_indices,
                                     p_values, f_values, p_bias, f_bias, SC);
  // 8) shared expert up: UP = x_ffn @ ffn_up_w
  k_gemm_bf16<<<dim3(2 * DSH / 64, SEQ2 / 128), 256, 0, stream>>>(
      Xffnb, DIM, ffn_up, 2 * DSH, UP, 2 * DSH, nullptr, DIM);
  // 9) H = silu(UP[:, :2048]) * UP[:, 2048:]
  k_silu_mul<<<(SEQ2 * DSH) / 256, 256, 0, stream>>>(UP, Hb);
  // 10) Y = H @ ffn_down_w + x_ffn_input   (initializes every output element)
  k_gemm_bf16<<<dim3(DIM / 64, SEQ2 / 128), 256, 0, stream>>>(
      Hb, DSH, ffn_down, DIM, Y, DIM, XFI, DSH);
  // 11) MoE grouped GEMM, atomic accumulate into Y
  k_moe<<<dim3(SEQ / 64, NEXP, 2), 256, 0, stream>>>(Xffnb, p_exp, f_exp,
                                                     p_indices, f_indices, SC, Y);
}
